// SpatialSelfAttention_14413910245916
// MI455X (gfx1250) — matrix operand order repaired and checked by reading, not compiled, not run on hardware
//
#include <hip/hip_runtime.h>
#include <hip/hip_bf16.h>

// ---------------------------------------------------------------------------
// SpatialSelfAttention for MI455X (gfx1250, wave32, WMMA bf16 w/ f32 accum)
// Pipeline: [cvt weights] -> conv(9 shifted GEMMs) -> qkv GEMM -> attention -> proj
// ---------------------------------------------------------------------------

#define DIMC   256
#define NHEADS 8
#define HD     32
#define WIN    8
#define NPOS   64          // WIN*WIN
#define DEPTH  100
#define BATCH  8
#define KCONV  (DIMC * 9)  // 2304
#define LDP    264         // padded LDS row stride (elems) for xT
#define ATT_SCALE 0.17677669529663687f  // 32^-0.5

typedef __attribute__((ext_vector_type(16))) __bf16 v16bf;
typedef __attribute__((ext_vector_type(8)))  __bf16 v8bf;
typedef __attribute__((ext_vector_type(8)))  float  v8f;

static __device__ inline v8f vzero8() {
  v8f z;
#pragma unroll
  for (int i = 0; i < 8; ++i) z[i] = 0.0f;
  return z;
}

static __device__ inline v8f wmma_bf16(v16bf a, v16bf b, v8f c) {
  // D = A(16x32) * B(32x16) + C(16x16), f32 accum
  return __builtin_amdgcn_wmma_f32_16x16x32_bf16(
      /*neg_a=*/false, a, /*neg_b=*/false, b,
      /*c_mod=*/(short)0, c, /*reuse_a=*/false, /*reuse_b=*/false);
}

// A fragment: A is MxK row-major bf16, row stride `ld` elems.
// A 16x32 frag order used here (not the ISA table order): lane (l&15) -> row; half=l>>4;
//   elems 0..15 = K = k0 + half*16 + {0..15} (contiguous, matches B frags)
static __device__ inline v16bf load_a_frag(const __bf16* A, int ld, int m0, int k0, int lane) {
  const int half = lane >> 4;
  const __bf16* p = A + (size_t)(m0 + (lane & 15)) * ld + k0 + half * 16;
  v8bf lo = *(const v8bf*)(p);
  v8bf hi = *(const v8bf*)(p + 8);
  v16bf f;
#pragma unroll
  for (int i = 0; i < 8; ++i) { f[i] = lo[i]; f[8 + i] = hi[i]; }
  return f;
}

// B fragment from a transposed ("Bt") matrix stored [n][k] with k contiguous,
// row stride `ld` elems. B 32x16 order used here (same as A, not ISA table): lane (l&15) -> column n; lanes
// 0-15 hold K=k0..k0+15, lanes 16-31 hold K=k0+16..k0+31 (contiguous runs).
static __device__ inline v16bf load_bt_frag(const __bf16* Bt, int ld, int n0, int k0, int lane) {
  const __bf16* p = Bt + (size_t)(n0 + (lane & 15)) * ld + k0 + (lane >> 4) * 16;
  v8bf lo = *(const v8bf*)(p);
  v8bf hi = *(const v8bf*)(p + 8);
  v16bf f;
#pragma unroll
  for (int i = 0; i < 8; ++i) { f[i] = lo[i]; f[8 + i] = hi[i]; }
  return f;
}

// B fragment where each lane already has its own row pointer (conv taps):
// rowp = &Bt[row(lane)][0]; reads k0 + half*16 .. +15 contiguously.
static __device__ inline v16bf load_row_bt_frag(const __bf16* rowp, int k0, int lane) {
  const __bf16* p = rowp + k0 + (lane >> 4) * 16;
  v8bf lo = *(const v8bf*)(p);
  v8bf hi = *(const v8bf*)(p + 8);
  v16bf f;
#pragma unroll
  for (int i = 0; i < 8; ++i) { f[i] = lo[i]; f[8 + i] = hi[i]; }
  return f;
}

// ---------------------------------------------------------------------------
// Stage 0a: plain f32 -> bf16 weight conversion (qkv_w, proj_w)
// ---------------------------------------------------------------------------
__global__ void ssa_cvt_bf16(const float* __restrict__ src, __bf16* __restrict__ dst, int n) {
  int i = blockIdx.x * blockDim.x + threadIdx.x;
  if (i < n) dst[i] = (__bf16)src[i];
}

// Stage 0b: conv weights (O=256, I=256, 1, 3, 3) -> bf16 re-laid as
// wA2[(o*9 + kid)*256 + cin] so each tap's A-matrix has cin contiguous.
__global__ void ssa_cvt_wconv(const float* __restrict__ src, __bf16* __restrict__ dst, int n) {
  int i = blockIdx.x * blockDim.x + threadIdx.x;
  if (i < n) {
    int o   = i / KCONV;
    int rem = i - o * KCONV;
    int kid = rem >> 8;        // rem / 256
    int ci  = rem & 255;       // rem % 256
    dst[i] = (__bf16)src[((size_t)o * DIMC + ci) * 9 + kid];
  }
}

// ---------------------------------------------------------------------------
// Stage 1: conv as 9 shifted GEMMs (C_out=256 x C_in=256 per tap, N=64).
// One workgroup per (b,d). x slice staged transposed in LDS: xT[10x10][cin],
// cin contiguous -> all B fragments are contiguous ds_load_b128 pairs.
// All 4 B fragments batched per k-step so loads clause + overlap the WMMAs.
// xl stored as [bd][pos][c] bf16 (K-contiguous for the QKV GEMM).
// ---------------------------------------------------------------------------
__global__ __launch_bounds__(256) void ssa_conv(
    const float* __restrict__ x, const __bf16* __restrict__ wA2,
    const float* __restrict__ convb, __bf16* __restrict__ xl) {
  __shared__ __bf16 xT[100 * LDP];  // zero-padded 10x10 spatial, cin contiguous

  const int bd = blockIdx.x;
  const int b = bd / DEPTH, d = bd % DEPTH;
  const int t = threadIdx.x, lane = t & 31, wv = t >> 5;
  const int half = lane >> 4;

  for (int i = t; i < 100 * LDP; i += 256) xT[i] = (__bf16)0.0f;
  __syncthreads();
  const float* xs = x + ((size_t)b * DIMC * DEPTH + d) * NPOS;
  for (int e = t; e < DIMC * NPOS; e += 256) {
    int c = e >> 6, pos = e & 63;               // consecutive lanes -> consecutive pos
    float v = xs[(size_t)c * (DEPTH * NPOS) + pos];
    int row = ((pos >> 3) + 1) * 10 + (pos & 7) + 1;
    xT[row * LDP + c] = (__bf16)v;
  }
  __syncthreads();

  const int m0 = wv * 32;
  // per-lane column position for each N tile
  int posn[4], pyn[4], pxn[4];
#pragma unroll
  for (int nj = 0; nj < 4; ++nj) {
    posn[nj] = nj * 16 + (lane & 15);
    pyn[nj] = posn[nj] >> 3;
    pxn[nj] = posn[nj] & 7;
  }

  v8f acc[2][4];
#pragma unroll
  for (int i = 0; i < 2; ++i)
#pragma unroll
    for (int j = 0; j < 4; ++j) acc[i][j] = vzero8();

  for (int kid = 0; kid < 9; ++kid) {
    const int ky = (kid >= 6) ? 2 : ((kid >= 3) ? 1 : 0);
    const int kx = kid - ky * 3;
    const __bf16* Atap = wA2 + kid * DIMC;  // ld = KCONV, offset selects this tap
    const __bf16* rowp[4];
#pragma unroll
    for (int nj = 0; nj < 4; ++nj)
      rowp[nj] = xT + ((pyn[nj] + ky) * 10 + pxn[nj] + kx) * LDP;

    for (int k0 = 0; k0 < DIMC; k0 += 32) {
      v16bf a0 = load_a_frag(Atap, KCONV, m0, k0, lane);
      v16bf a1 = load_a_frag(Atap, KCONV, m0 + 16, k0, lane);
      v16bf bfr[4];
#pragma unroll
      for (int nj = 0; nj < 4; ++nj) bfr[nj] = load_row_bt_frag(rowp[nj], k0, lane);
#pragma unroll
      for (int nj = 0; nj < 4; ++nj) {
        acc[0][nj] = wmma_bf16(a0, bfr[nj], acc[0][nj]);
        acc[1][nj] = wmma_bf16(a1, bfr[nj], acc[1][nj]);
      }
    }
  }

  __bf16* dst = xl + (size_t)bd * NPOS * DIMC;
#pragma unroll
  for (int mi = 0; mi < 2; ++mi)
#pragma unroll
    for (int nj = 0; nj < 4; ++nj) {
      const int pos = posn[nj];
#pragma unroll
      for (int r = 0; r < 8; ++r) {
        int m = m0 + mi * 16 + r + 8 * half;
        dst[(size_t)pos * DIMC + m] = (__bf16)(acc[mi][nj][r] + convb[m]);
      }
    }
}

// ---------------------------------------------------------------------------
// Stage 2: QKV GEMM (768x256 @ 256x64 per (b,d)), split by grid.y = section s.
// q,k stored [h][pos][c] (transposed, for QK^T A/B frags); v stored [h][c][pos].
// ---------------------------------------------------------------------------
__global__ __launch_bounds__(256) void ssa_qkv(
    const __bf16* __restrict__ wQ, const float* __restrict__ qkvb,
    const __bf16* __restrict__ xl, __bf16* __restrict__ qkv) {
  const int bd = blockIdx.x, s = blockIdx.y;
  const int t = threadIdx.x, lane = t & 31, wv = t >> 5;
  const int half = lane >> 4;
  const int m0 = wv * 32;

  const __bf16* A  = wQ + (size_t)s * DIMC * DIMC;   // 256 rows of this section
  const __bf16* Bt = xl + (size_t)bd * NPOS * DIMC;  // [pos][c]

  v8f acc[2][4];
#pragma unroll
  for (int i = 0; i < 2; ++i)
#pragma unroll
    for (int j = 0; j < 4; ++j) acc[i][j] = vzero8();

#pragma unroll
  for (int k0 = 0; k0 < DIMC; k0 += 32) {
    v16bf a0 = load_a_frag(A, DIMC, m0, k0, lane);
    v16bf a1 = load_a_frag(A, DIMC, m0 + 16, k0, lane);
    v16bf bfr[4];
#pragma unroll
    for (int nj = 0; nj < 4; ++nj) bfr[nj] = load_bt_frag(Bt, DIMC, nj * 16, k0, lane);
#pragma unroll
    for (int nj = 0; nj < 4; ++nj) {
      acc[0][nj] = wmma_bf16(a0, bfr[nj], acc[0][nj]);
      acc[1][nj] = wmma_bf16(a1, bfr[nj], acc[1][nj]);
    }
  }

  __bf16* base = qkv + (size_t)(bd * 3 + s) * (NHEADS * HD * NPOS);
#pragma unroll
  for (int mi = 0; mi < 2; ++mi)
#pragma unroll
    for (int nj = 0; nj < 4; ++nj) {
      const int pos = nj * 16 + (lane & 15);
#pragma unroll
      for (int r = 0; r < 8; ++r) {
        int m = m0 + mi * 16 + r + 8 * half;
        float v = acc[mi][nj][r] + qkvb[s * DIMC + m];
        int h = m >> 5, c = m & 31;
        size_t idx = (s < 2) ? (((size_t)h * NPOS + pos) * HD + c)   // [h][pos][c]
                             : (((size_t)h * HD + c) * NPOS + pos);  // [h][c][pos]
        base[idx] = (__bf16)v;
      }
    }
}

// ---------------------------------------------------------------------------
// Stage 3: attention. One workgroup per (b,d); one wave per head.
// QK^T (64x64x32) -> fp32 softmax + rel-pos bias -> A*V (32x64x64).
// xa stored [bd][pos][channel] bf16 for the proj GEMM.
// ---------------------------------------------------------------------------
__global__ __launch_bounds__(256) void ssa_attn(
    const __bf16* __restrict__ qkv, const float* __restrict__ rpb,
    __bf16* __restrict__ xa) {
  extern __shared__ char smem[];
  float* attn = (float*)smem;                              // [8][64][64] f32
  float* rpbs = (float*)(smem + NHEADS * NPOS * NPOS * 4); // [225][8]

  const int bd = blockIdx.x;
  const int t = threadIdx.x, lane = t & 31, h = t >> 5;
  const int half = lane >> 4;

  for (int i = t; i < (2 * WIN - 1) * (2 * WIN - 1) * NHEADS; i += 256) rpbs[i] = rpb[i];
  __syncthreads();

  const __bf16* qt = qkv + ((size_t)(bd * 3 + 0) * NHEADS + h) * (NPOS * HD); // [pos][c]
  const __bf16* kt = qkv + ((size_t)(bd * 3 + 1) * NHEADS + h) * (NPOS * HD); // [pos][c]
  const __bf16* vv = qkv + ((size_t)(bd * 3 + 2) * NHEADS + h) * (HD * NPOS); // [c][pos]
  float* attnh = attn + (size_t)h * NPOS * NPOS;

  // --- logits = (q^T k) * scale ---
  {
    v16bf afr[4], bfr[4];
#pragma unroll
    for (int i = 0; i < 4; ++i) {
      afr[i] = load_a_frag(qt, HD, i * 16, 0, lane);   // A[n][c]
      bfr[i] = load_bt_frag(kt, HD, i * 16, 0, lane);  // B[c][m] via kt[m][c]
    }
#pragma unroll
    for (int ni = 0; ni < 4; ++ni)
#pragma unroll
      for (int mj = 0; mj < 4; ++mj) {
        v8f a = wmma_bf16(afr[ni], bfr[mj], vzero8());
#pragma unroll
        for (int r = 0; r < 8; ++r) {
          int n = ni * 16 + r + 8 * half;
          int m = mj * 16 + (lane & 15);
          attnh[n * NPOS + m] = a[r] * ATT_SCALE;
        }
      }
  }

  // --- softmax rows (wave-local: lanes cover all 64 rows of this head) ---
  // In-place repack to bf16: row n occupies first 128B of the 256B f32 row.
  __bf16* attnb = (__bf16*)attnh;  // row stride 128 bf16 elems
  for (int rr = 0; rr < 2; ++rr) {
    const int n = lane + rr * 32;
    const int i1 = n >> 3, j1 = n & 7;
    float row[NPOS];
    float mx = -1e30f;
#pragma unroll
    for (int m = 0; m < NPOS; ++m) {
      int i2 = m >> 3, j2 = m & 7;
      int rel = (i1 - i2 + 7) * 15 + (j1 - j2 + 7);
      float v = attnh[n * NPOS + m] + rpbs[rel * NHEADS + h];
      row[m] = v;
      mx = fmaxf(mx, v);
    }
    float sum = 0.0f;
#pragma unroll
    for (int m = 0; m < NPOS; ++m) { row[m] = __expf(row[m] - mx); sum += row[m]; }
    float inv = 1.0f / sum;
#pragma unroll
    for (int m = 0; m < NPOS; ++m) attnb[n * 128 + m] = (__bf16)(row[m] * inv);
  }

  // --- xa = v(32x64) * attn^T(64x64) ---
  v8f acc[2][4];
#pragma unroll
  for (int i = 0; i < 2; ++i)
#pragma unroll
    for (int j = 0; j < 4; ++j) acc[i][j] = vzero8();

#pragma unroll
  for (int kk = 0; kk < 2; ++kk) {
    const int k0 = kk * 32;
    v16bf a0 = load_a_frag(vv, NPOS, 0, k0, lane);
    v16bf a1 = load_a_frag(vv, NPOS, 16, k0, lane);
    v16bf bfr[4];
#pragma unroll
    for (int nj = 0; nj < 4; ++nj)
      bfr[nj] = load_bt_frag(attnb, 128, nj * 16, k0, lane);  // B[m][n] via attn[n][m]
#pragma unroll
    for (int nj = 0; nj < 4; ++nj) {
      acc[0][nj] = wmma_bf16(a0, bfr[nj], acc[0][nj]);
      acc[1][nj] = wmma_bf16(a1, bfr[nj], acc[1][nj]);
    }
  }

  __bf16* dst = xa + (size_t)bd * NPOS * DIMC;
#pragma unroll
  for (int mi = 0; mi < 2; ++mi)
#pragma unroll
    for (int nj = 0; nj < 4; ++nj) {
      const int pos = nj * 16 + (lane & 15);
#pragma unroll
      for (int r = 0; r < 8; ++r) {
        int c = mi * 16 + r + 8 * half;
        dst[(size_t)pos * DIMC + h * HD + c] = (__bf16)acc[mi][nj][r];
      }
    }
}

// ---------------------------------------------------------------------------
// Stage 4: output projection (256x256 @ 256x64 per (b,d)) + bias, f32 out.
// ---------------------------------------------------------------------------
__global__ __launch_bounds__(256) void ssa_proj(
    const __bf16* __restrict__ wP, const float* __restrict__ projb,
    const __bf16* __restrict__ xa, float* __restrict__ out) {
  const int bd = blockIdx.x;
  const int b = bd / DEPTH, d = bd % DEPTH;
  const int t = threadIdx.x, lane = t & 31, wv = t >> 5;
  const int half = lane >> 4;
  const int m0 = wv * 32;

  const __bf16* Bt = xa + (size_t)bd * NPOS * DIMC;

  v8f acc[2][4];
#pragma unroll
  for (int i = 0; i < 2; ++i)
#pragma unroll
    for (int j = 0; j < 4; ++j) acc[i][j] = vzero8();

#pragma unroll
  for (int k0 = 0; k0 < DIMC; k0 += 32) {
    v16bf a0 = load_a_frag(wP, DIMC, m0, k0, lane);
    v16bf a1 = load_a_frag(wP, DIMC, m0 + 16, k0, lane);
    v16bf bfr[4];
#pragma unroll
    for (int nj = 0; nj < 4; ++nj) bfr[nj] = load_bt_frag(Bt, DIMC, nj * 16, k0, lane);
#pragma unroll
    for (int nj = 0; nj < 4; ++nj) {
      acc[0][nj] = wmma_bf16(a0, bfr[nj], acc[0][nj]);
      acc[1][nj] = wmma_bf16(a1, bfr[nj], acc[1][nj]);
    }
  }

#pragma unroll
  for (int mi = 0; mi < 2; ++mi)
#pragma unroll
    for (int nj = 0; nj < 4; ++nj) {
      const int pos = nj * 16 + (lane & 15);
#pragma unroll
      for (int r = 0; r < 8; ++r) {
        int m = m0 + mi * 16 + r + 8 * half;
        out[(((size_t)(b * DIMC + m)) * DEPTH + d) * NPOS + pos] =
            acc[mi][nj][r] + projb[m];
      }
    }
}

// ---------------------------------------------------------------------------
// Host-side launch
// ---------------------------------------------------------------------------
extern "C" void kernel_launch(void* const* d_in, const int* in_sizes, int n_in,
                              void* d_out, int out_size, void* d_ws, size_t ws_size,
                              hipStream_t stream) {
  (void)in_sizes; (void)n_in; (void)out_size; (void)ws_size;
  const float* x        = (const float*)d_in[0];
  const float* dwconv_w = (const float*)d_in[1];
  const float* dwconv_b = (const float*)d_in[2];
  const float* qkv_w    = (const float*)d_in[3];
  const float* qkv_b    = (const float*)d_in[4];
  const float* proj_w   = (const float*)d_in[5];
  const float* proj_b   = (const float*)d_in[6];
  const float* rpb      = (const float*)d_in[7];
  float* out = (float*)d_out;

  char* ws = (char*)d_ws;
  const size_t nWA = (size_t)DIMC * KCONV;          // 589,824
  const size_t nWQ = (size_t)3 * DIMC * DIMC;       // 196,608
  const size_t nWP = (size_t)DIMC * DIMC;           //  65,536
  const size_t nBD = (size_t)BATCH * DEPTH;         // 800

  __bf16* wA2 = (__bf16*)(ws);
  __bf16* wQ  = (__bf16*)(ws + 1179648);
  __bf16* wP  = (__bf16*)(ws + 1572864);
  __bf16* xl  = (__bf16*)(ws + 1703936);
  __bf16* qkv = (__bf16*)(ws + 27918336);
  __bf16* xa  = (__bf16*)(ws + 106561536);

  ssa_cvt_wconv<<<(int)((nWA + 255) / 256), 256, 0, stream>>>(dwconv_w, wA2, (int)nWA);
  ssa_cvt_bf16<<<(int)((nWQ + 255) / 256), 256, 0, stream>>>(qkv_w, wQ, (int)nWQ);
  ssa_cvt_bf16<<<(int)((nWP + 255) / 256), 256, 0, stream>>>(proj_w, wP, (int)nWP);

  ssa_conv<<<(int)nBD, 256, 0, stream>>>(x, wA2, dwconv_b, xl);
  ssa_qkv<<<dim3((int)nBD, 3), 256, 0, stream>>>(wQ, qkv_b, xl, qkv);

  const size_t smem_attn = (size_t)NHEADS * NPOS * NPOS * 4 + 225 * NHEADS * 4; // 138,272 B
  ssa_attn<<<(int)nBD, 256, smem_attn, stream>>>(qkv, rpb, xa);

  ssa_proj<<<(int)nBD, 256, 0, stream>>>(wP, proj_b, xa, out);
}
